// SelfAttention_36816459661846
// MI455X (gfx1250) — matrix operand order repaired and checked by reading, not compiled, not run on hardware
//
#include <hip/hip_runtime.h>
#include <stdint.h>

#define DIM     1024
#define NHEADS  16
#define HDIM    64
#define SEQ     4096
#define BATCH   2
#define ROWS    (BATCH * SEQ)    // 8192
#define QKVC    (3 * DIM)        // 3072
#define BH      (BATCH * NHEADS) // 32
#define NIT     (SEQ / 32)       // 128 key tiles

typedef __attribute__((ext_vector_type(16))) __bf16 v16bf;
typedef __attribute__((ext_vector_type(8)))  float  f32x8;
typedef int i32x4v __attribute__((vector_size(16)));

union Frag { v16bf v; unsigned u[8]; };

#ifndef __has_builtin
#define __has_builtin(x) 0
#endif
#if __has_builtin(__builtin_amdgcn_global_load_async_to_lds_b128)
#define HAVE_ASYNC_LDS 1
#else
#define HAVE_ASYNC_LDS 0
#endif

#if HAVE_ASYNC_LDS
#if __has_builtin(__builtin_amdgcn_s_wait_asynccnt)
#define WAIT_ASYNC(n) __builtin_amdgcn_s_wait_asynccnt(n)
#else
#define WAIT_ASYNC(n) asm volatile("s_wait_asynccnt " #n ::: "memory")
#endif
#else
#define WAIT_ASYNC(n) do {} while (0)
#endif

// Copy 16 bytes global -> LDS (async DMA; ASYNCcnt-tracked).
__device__ __forceinline__ void cp16_g2l(const unsigned* g, unsigned* l) {
#if HAVE_ASYNC_LDS
  __builtin_amdgcn_global_load_async_to_lds_b128(
      (__attribute__((address_space(1))) i32x4v*)g,
      (__attribute__((address_space(3))) i32x4v*)l, 0, 0);
#else
  *(uint4*)l = *(const uint4*)g;
#endif
}

__device__ __forceinline__ unsigned short f2bf(float f) {
  unsigned u = __float_as_uint(f);
  u += 0x7fffu + ((u >> 16) & 1u);      // round-to-nearest-even
  return (unsigned short)(u >> 16);
}

// Pack two f32 -> packed bf16 dword; single v_cvt_pk_bf16_f32 when available.
#if __has_builtin(__builtin_amdgcn_cvt_pk_bf16_f32)
__device__ __forceinline__ unsigned packbf(float lo, float hi) {
  auto t = __builtin_amdgcn_cvt_pk_bf16_f32(lo, hi);
  unsigned r;
  __builtin_memcpy(&r, &t, sizeof(r));
  return r;
}
#else
__device__ __forceinline__ unsigned packbf(float lo, float hi) {
  return (unsigned)f2bf(lo) | ((unsigned)f2bf(hi) << 16);
}
#endif

// A-operand (16x32 bf16) per-lane dword(pair) index inside a 32-wide K chunk.
// Lanes 0-15: K 0..7 (v0..3), 16..23 (v4..7). Lanes 16-31: +8.
__device__ __forceinline__ int apair(int v, int hi) {
  return ((v < 4) ? v : (v + 4)) + (hi ? 4 : 0);
}
// B-operand (32x16 bf16): same K order as A-operand.
__device__ __forceinline__ int bpair(int v, int hi) { return apair(v, hi); }

__device__ __forceinline__ f32x8 wmma_bf16(const Frag& a, const Frag& b, f32x8 c) {
  return __builtin_amdgcn_wmma_f32_16x16x32_bf16(false, a.v, false, b.v,
                                                 (short)0, c, false, false);
}

// ---------------------------------------------------------------------------
// Repack fp32 weights [K, N] -> bf16 pair-packed dwords [K/2, N]
// ---------------------------------------------------------------------------
__global__ void pack_bf16_pairs(const float* __restrict__ W,
                                unsigned* __restrict__ Wp, int K2, int N) {
  int i = blockIdx.x * blockDim.x + threadIdx.x;
  if (i >= K2 * N) return;
  int k2 = i / N, n = i - k2 * N;
  Wp[i] = packbf(W[(size_t)(2 * k2) * N + n], W[(size_t)(2 * k2 + 1) * N + n]);
}

// ---------------------------------------------------------------------------
// QKV GEMM: [8192,1024]f32 @ [1024,3072]bf16p + bias -> scattered bf16 Q/K/Vp
// One wave computes one 16x16 tile. Q,K: [bh, n, d]. V: seq-pair-packed.
// ---------------------------------------------------------------------------
__global__ __launch_bounds__(32) void qkv_gemm(
    const float* __restrict__ X, const unsigned* __restrict__ Wp,
    const float* __restrict__ bias,
    unsigned short* __restrict__ Q, unsigned short* __restrict__ Kb,
    unsigned short* __restrict__ Vp) {
  const int lane = threadIdx.x;
  const int hi = lane >> 4, l16 = lane & 15;
  const int row0 = blockIdx.y << 4, col0 = blockIdx.x << 4;
  f32x8 c = {0.f, 0.f, 0.f, 0.f, 0.f, 0.f, 0.f, 0.f};
  const float*    xrow = X + (size_t)(row0 + l16) * DIM;
  const unsigned* wcol = Wp + col0 + l16;
  for (int kk = 0; kk < DIM; kk += 32) {
    __builtin_prefetch((const void*)(xrow + kk + 64), 0, 0);
    Frag a, b;
#pragma unroll
    for (int v = 0; v < 8; ++v) {
      int p = apair(v, hi);
      float2 t = *(const float2*)(xrow + kk + 2 * p);
      a.u[v] = packbf(t.x, t.y);
    }
#pragma unroll
    for (int v = 0; v < 8; ++v)
      b.u[v] = wcol[(size_t)((kk >> 1) + bpair(v, hi)) * QKVC];
    c = wmma_bf16(a, b, c);
  }
  const int col = col0 + l16;
  const float bv = bias[col];
  const int which = col >> 10;            // 0=Q 1=K 2=V (uniform per wave)
  const int cc = col & (DIM - 1);
  const int h = cc >> 6, d = cc & 63;
#pragma unroll
  for (int r = 0; r < 8; ++r) {
    int row = row0 + r + (hi << 3);
    int bb = row >> 12, n = row & (SEQ - 1);
    int bh = (bb << 4) + h;
    unsigned short val = f2bf(c[r] + bv);
    if (which == 0)
      Q[((size_t)bh * SEQ + n) * HDIM + d] = val;
    else if (which == 1)
      Kb[((size_t)bh * SEQ + n) * HDIM + d] = val;
    else
      Vp[((size_t)bh * (SEQ / 2) + (n >> 1)) * (2 * HDIM) + 2 * d + (n & 1)] = val;
  }
}

// ---------------------------------------------------------------------------
// Flash attention: 4 waves/block, each wave owns one 16-row q tile.
// K/V tiles (contiguous 4KB each) are async-DMA'd into double-buffered LDS
// and shared by all 4 waves. Online softmax in fp32; P transposed to
// A-layout through per-wave LDS scratch.
// ---------------------------------------------------------------------------
__global__ __launch_bounds__(128) void flash_attn(
    const unsigned short* __restrict__ Q, const unsigned short* __restrict__ Kb,
    const unsigned* __restrict__ Vp, unsigned short* __restrict__ O) {
  const int tid = threadIdx.x;
  const int lane = tid & 31, wave = tid >> 5;
  const int hi = (lane >> 4) & 1, l16 = lane & 15;
  const int bh = blockIdx.y;
  const int q0 = (blockIdx.x * 4 + wave) << 4;
  const float scale = 0.125f;   // 1/sqrt(64)

  __shared__ unsigned sK[2][32 * 32];  // 32 key rows x 64 bf16   (4KB each)
  __shared__ unsigned sV[2][16 * 64];  // 16 pair rows x 64 dwords (4KB each)
  __shared__ unsigned sP[4][16 * 16];  // per-wave P transpose scratch

  const unsigned* kg = (const unsigned*)Kb + (size_t)bh * SEQ * 32;
  const unsigned* vg = Vp + (size_t)bh * (SEQ / 2) * HDIM;

  // Cooperative stage of key-tile t into buffer buf (128 threads x 2 x 16B x2)
  auto stage = [&](int t, int buf) {
#pragma unroll
    for (int c = 0; c < 2; ++c) {
      int ch4 = (tid + c * 128) * 4;
      cp16_g2l(kg + (size_t)t * 1024 + ch4, &sK[buf][ch4]);
      cp16_g2l(vg + (size_t)t * 1024 + ch4, &sV[buf][ch4]);
    }
  };
  stage(0, 0);
  stage(1, 1);

  // Persistent Q A-fragments for both 32-wide d chunks
  const unsigned* qrow = (const unsigned*)Q + ((size_t)bh * SEQ + q0 + l16) * 32;
  Frag qf[2];
#pragma unroll
  for (int c2 = 0; c2 < 2; ++c2)
#pragma unroll
    for (int v = 0; v < 8; ++v) qf[c2].u[v] = qrow[c2 * 16 + apair(v, hi)];

  f32x8 o0 = {0.f,0.f,0.f,0.f,0.f,0.f,0.f,0.f};
  f32x8 o1 = o0, o2 = o0, o3 = o0;
  float mrow[8], lrow[8];
#pragma unroll
  for (int r = 0; r < 8; ++r) { mrow[r] = -3.0e38f; lrow[r] = 0.f; }

  for (int it = 0; it < NIT; ++it) {
    const int cur = it & 1;
    if (it == NIT - 1) { WAIT_ASYNC(0); } else { WAIT_ASYNC(4); }
    __syncthreads();                       // tile `cur` resident in LDS
    const unsigned* kl = sK[cur];
    const unsigned* vl = sV[cur];

    f32x8 s0 = {0.f,0.f,0.f,0.f,0.f,0.f,0.f,0.f};
    f32x8 s1 = s0;
#pragma unroll
    for (int c2 = 0; c2 < 2; ++c2) {
      Frag kf;
#pragma unroll
      for (int v = 0; v < 8; ++v) kf.u[v] = kl[(size_t)l16 * 32 + c2 * 16 + bpair(v, hi)];
      s0 = wmma_bf16(qf[c2], kf, s0);
    }
#pragma unroll
    for (int c2 = 0; c2 < 2; ++c2) {
      Frag kf;
#pragma unroll
      for (int v = 0; v < 8; ++v) kf.u[v] = kl[(size_t)(16 + l16) * 32 + c2 * 16 + bpair(v, hi)];
      s1 = wmma_bf16(qf[c2], kf, s1);
    }

    // Online softmax: C/D register r is row (r + hi*8), col = l16.
    unsigned short* shb = (unsigned short*)sP[wave];
#pragma unroll
    for (int r = 0; r < 8; ++r) {
      float v0 = s0[r] * scale, v1 = s1[r] * scale;
      float mx = fmaxf(v0, v1);
#pragma unroll
      for (int off = 1; off < 16; off <<= 1)
        mx = fmaxf(mx, __shfl_xor(mx, off, 32));
      float mn = fmaxf(mrow[r], mx);
      float alpha = __expf(mrow[r] - mn);
      mrow[r] = mn;
      float p0 = __expf(v0 - mn), p1 = __expf(v1 - mn);
      float sum = p0 + p1;
#pragma unroll
      for (int off = 1; off < 16; off <<= 1) sum += __shfl_xor(sum, off, 32);
      lrow[r] = lrow[r] * alpha + sum;
      o0[r] *= alpha; o1[r] *= alpha; o2[r] *= alpha; o3[r] *= alpha;
      unsigned short* shrow = shb + (r + (hi << 3)) * 32;
      shrow[l16]      = f2bf(p0);   // key cols 0..15
      shrow[16 + l16] = f2bf(p1);   // key cols 16..31
    }
    asm volatile("s_wait_dscnt 0x0" ::: "memory");  // D-layout -> A-layout

    Frag pf;
#pragma unroll
    for (int v = 0; v < 8; ++v) pf.u[v] = sP[wave][l16 * 16 + apair(v, hi)];

    Frag vf;
#pragma unroll
    for (int v = 0; v < 8; ++v) vf.u[v] = vl[(size_t)bpair(v, hi) * 64 + 0 + l16];
    o0 = wmma_bf16(pf, vf, o0);
#pragma unroll
    for (int v = 0; v < 8; ++v) vf.u[v] = vl[(size_t)bpair(v, hi) * 64 + 16 + l16];
    o1 = wmma_bf16(pf, vf, o1);
#pragma unroll
    for (int v = 0; v < 8; ++v) vf.u[v] = vl[(size_t)bpair(v, hi) * 64 + 32 + l16];
    o2 = wmma_bf16(pf, vf, o2);
#pragma unroll
    for (int v = 0; v < 8; ++v) vf.u[v] = vl[(size_t)bpair(v, hi) * 64 + 48 + l16];
    o3 = wmma_bf16(pf, vf, o3);

    __syncthreads();                       // everyone done reading tile `cur`
    if (it + 2 < NIT) stage(it + 2, cur);  // refill freed buffer (async)
  }

  // Normalize and write O as bf16 [B*N, D] (row-major, ready as next A operand)
  const int b = bh >> 4, h = bh & 15;
#pragma unroll
  for (int r = 0; r < 8; ++r) {
    float inv = 1.f / lrow[r];
    size_t base = ((size_t)b * SEQ + q0 + r + (hi << 3)) * DIM + h * HDIM + l16;
    O[base + 0]  = f2bf(o0[r] * inv);
    O[base + 16] = f2bf(o1[r] * inv);
    O[base + 32] = f2bf(o2[r] * inv);
    O[base + 48] = f2bf(o3[r] * inv);
  }
}

// ---------------------------------------------------------------------------
// Output projection: bf16 [8192,1024] @ bf16p [1024,1024] + bias -> f32
// ---------------------------------------------------------------------------
__global__ __launch_bounds__(32) void out_proj(
    const unsigned short* __restrict__ A, const unsigned* __restrict__ Wp,
    const float* __restrict__ bias, float* __restrict__ Cout) {
  const int lane = threadIdx.x, hi = lane >> 4, l16 = lane & 15;
  const int row0 = blockIdx.y << 4, col0 = blockIdx.x << 4;
  f32x8 c = {0.f, 0.f, 0.f, 0.f, 0.f, 0.f, 0.f, 0.f};
  const unsigned* arow = (const unsigned*)A + (size_t)(row0 + l16) * (DIM / 2);
  const unsigned* wcol = Wp + col0 + l16;
  for (int kk = 0; kk < DIM; kk += 32) {
    __builtin_prefetch((const void*)(arow + (kk >> 1) + 32), 0, 0);
    Frag a, b;
#pragma unroll
    for (int v = 0; v < 8; ++v) a.u[v] = arow[(kk >> 1) + apair(v, hi)];
#pragma unroll
    for (int v = 0; v < 8; ++v)
      b.u[v] = wcol[(size_t)((kk >> 1) + bpair(v, hi)) * DIM];
    c = wmma_bf16(a, b, c);
  }
  const float bv = bias[col0 + l16];
#pragma unroll
  for (int r = 0; r < 8; ++r)
    Cout[(size_t)(row0 + r + (hi << 3)) * DIM + col0 + l16] = c[r] + bv;
}

// ---------------------------------------------------------------------------
extern "C" void kernel_launch(void* const* d_in, const int* in_sizes, int n_in,
                              void* d_out, int out_size, void* d_ws, size_t ws_size,
                              hipStream_t stream) {
  const float* x     = (const float*)d_in[0];
  const float* w_qkv = (const float*)d_in[1];
  const float* b_qkv = (const float*)d_in[2];
  const float* w_o   = (const float*)d_in[3];
  const float* b_o   = (const float*)d_in[4];
  float* out = (float*)d_out;

  char* ws = (char*)d_ws;
  size_t off = 0;
  unsigned short* Q  = (unsigned short*)(ws + off); off += (size_t)BH * SEQ * HDIM * 2;
  unsigned short* Kb = (unsigned short*)(ws + off); off += (size_t)BH * SEQ * HDIM * 2;
  unsigned short* Vp = (unsigned short*)(ws + off); off += (size_t)BH * SEQ * HDIM * 2;
  unsigned short* O  = (unsigned short*)(ws + off); off += (size_t)ROWS * DIM * 2;
  unsigned* WqkvP    = (unsigned*)(ws + off);       off += (size_t)(DIM / 2) * QKVC * 4;
  unsigned* WoP      = (unsigned*)(ws + off);       off += (size_t)(DIM / 2) * DIM * 4;
  (void)ws_size; (void)in_sizes; (void)n_in; (void)out_size;

  int tot1 = (DIM / 2) * QKVC;
  pack_bf16_pairs<<<(tot1 + 255) / 256, 256, 0, stream>>>(w_qkv, WqkvP, DIM / 2, QKVC);
  int tot2 = (DIM / 2) * DIM;
  pack_bf16_pairs<<<(tot2 + 255) / 256, 256, 0, stream>>>(w_o, WoP, DIM / 2, DIM);

  qkv_gemm<<<dim3(QKVC / 16, ROWS / 16), 32, 0, stream>>>(x, WqkvP, b_qkv, Q, Kb, Vp);
  flash_attn<<<dim3(SEQ / 64, BH), 128, 0, stream>>>(Q, Kb, (const unsigned*)Vp, O);
  out_proj<<<dim3(DIM / 16, ROWS / 16), 32, 0, stream>>>(O, WoP, b_o, out);
}
